// SpatialTemporal_MS_GCN_6837587935981
// MI455X (gfx1250) — matrix operand order repaired and checked by reading, not compiled, not run on hardware
//
#include <hip/hip_runtime.h>

typedef __attribute__((ext_vector_type(2)))  float    v2f;
typedef __attribute__((ext_vector_type(8)))  float    v8f;
typedef __attribute__((ext_vector_type(16))) _Float16 v16h;

#define N_   32
#define C_   3
#define T_   300
#define V_   25
#define OUT_ 96
#define VP   32      // 25 padded to 32
#define TCH  20      // t-values per block in Asum kernel (300/20 = 15 chunks)
#define TBLK 15      // t-values per block in fused kernel (300/15 = 20 chunks)

// ---------------- kernel 0: zero the A_sum workspace ----------------
__global__ void k_zero_ws(float* ws, int n) {
  int i = blockIdx.x * blockDim.x + threadIdx.x;
  if (i < n) ws[i] = 0.f;
}

// ---- kernel 1: per (n, t-chunk): build A, A^2, A^3 per t, accumulate locally,
// ---- one atomic pass per block. grid = N_*(T_/TCH), block = 128 (4 waves).
// ---- Asum layout: [n][s=1..3][25][25]
__global__ void k_build_Asum(const float* __restrict__ xx, float* __restrict__ Asum) {
  __shared__ float xs[C_ * V_];
  __shared__ float As [VP * VP];
  __shared__ float A2s[VP * VP];
  __shared__ float A3s[VP * VP];
  __shared__ float Ssum[3 * V_ * V_];   // local temporal accumulator

  const int n  = blockIdx.x / (T_ / TCH);
  const int tc = blockIdx.x % (T_ / TCH);
  const int tid = threadIdx.x;

  // each thread owns fixed Ssum indices (tid, tid+128, ...) -> no races, no barriers
  for (int i = tid; i < 3 * V_ * V_; i += 128) Ssum[i] = 0.f;

  const int wave = tid >> 5;
  const int lane = tid & 31;
  const int hi   = lane >> 4;          // upper half-wave
  const int lm   = lane & 15;
  const int mrow = (wave >> 1) * 16;   // 2x2 tile grid over 32x32
  const int ncol = (wave & 1) * 16;

  for (int tt = 0; tt < TCH; ++tt) {
    const int t = tc * TCH + tt;

    // xs[c*25+v] = xx[n,c,t,v,0]
    for (int i = tid; i < C_ * V_; i += 128) {
      int c = i / V_, v = i % V_;
      xs[i] = xx[((n * C_ + c) * T_ + t) * V_ + v];
    }
    __syncthreads();   // also fences previous iteration's Ssum accumulation

    // A[r,c] = exp(-||x_r - x_c||^2), zero-padded to 32x32
    for (int i = tid; i < VP * VP; i += 128) {
      int r = i >> 5, cc = i & 31;
      float val = 0.f;
      if (r < V_ && cc < V_) {
        float d2 = 0.f;
        #pragma unroll
        for (int c = 0; c < C_; ++c) {
          float d = xs[c * V_ + r] - xs[c * V_ + cc];
          d2 += d * d;
        }
        val = __expf(-d2);
      }
      As[i] = val;
    }
    __syncthreads();

#if __has_builtin(__builtin_amdgcn_wmma_f32_16x16x4_f32)
    // --- A2 = A @ A (f32 WMMA, K stepped by 4 over padded 32) ---
    {
      v8f acc = {};
      #pragma unroll
      for (int k0 = 0; k0 < VP; k0 += 4) {
        int ka = k0 + 2 * hi;            // lanes 16-31 carry K+2,K+3
        v2f a, b;
        a.x = As[(mrow + lm) * VP + ka];
        a.y = As[(mrow + lm) * VP + ka + 1];
        b.x = As[ka * VP + ncol + lm];
        b.y = As[(ka + 1) * VP + ncol + lm];
        acc = __builtin_amdgcn_wmma_f32_16x16x4_f32(false, a, false, b,
                                                    (short)0, acc, false, false);
      }
      #pragma unroll
      for (int r = 0; r < 8; ++r)
        A2s[(mrow + 8 * hi + r) * VP + ncol + lm] = acc[r];
    }
    __syncthreads();
    // --- A3 = A2 @ A ---
    {
      v8f acc = {};
      #pragma unroll
      for (int k0 = 0; k0 < VP; k0 += 4) {
        int ka = k0 + 2 * hi;
        v2f a, b;
        a.x = A2s[(mrow + lm) * VP + ka];
        a.y = A2s[(mrow + lm) * VP + ka + 1];
        b.x = As[ka * VP + ncol + lm];
        b.y = As[(ka + 1) * VP + ncol + lm];
        acc = __builtin_amdgcn_wmma_f32_16x16x4_f32(false, a, false, b,
                                                    (short)0, acc, false, false);
      }
      #pragma unroll
      for (int r = 0; r < 8; ++r)
        A3s[(mrow + 8 * hi + r) * VP + ncol + lm] = acc[r];
    }
    __syncthreads();
#else
    for (int i = tid; i < V_ * V_; i += 128) {
      int r = i / V_, cc = i % V_;
      float acc = 0.f;
      for (int u = 0; u < V_; ++u) acc += As[r * VP + u] * As[u * VP + cc];
      A2s[r * VP + cc] = acc;
    }
    __syncthreads();
    for (int i = tid; i < V_ * V_; i += 128) {
      int r = i / V_, cc = i % V_;
      float acc = 0.f;
      for (int u = 0; u < V_; ++u) acc += A2s[r * VP + u] * As[u * VP + cc];
      A3s[r * VP + cc] = acc;
    }
    __syncthreads();
#endif

    // local accumulate [A; A2; A3] (same thread -> same indices every iter)
    for (int i = tid; i < 3 * V_ * V_; i += 128) {
      int s = i / (V_ * V_);
      int rem = i % (V_ * V_);
      int r = rem / V_, cc = rem % V_;
      const float* src = (s == 0) ? As : (s == 1) ? A2s : A3s;
      Ssum[i] += src[r * VP + cc];
    }
    __syncthreads();
  }

  // single atomic pass per block (identity block folded in consumer)
  float* dst = Asum + n * (3 * V_ * V_);
  for (int i = tid; i < 3 * V_ * V_; i += 128)
    atomicAdd(&dst[i], Ssum[i]);
}

// ---------------- kernel 2: fused agg + 1x1 conv + BN + ReLU ----------------
// grid = N_*(T_/TBLK), block = 256 (8 waves)
__global__ void k_fused_out(const float* __restrict__ x, const float* __restrict__ W,
                            const float* __restrict__ bias, const float* __restrict__ gamma,
                            const float* __restrict__ beta, const float* __restrict__ Asum,
                            float* __restrict__ out) {
  __shared__ float    As[3 * V_ * V_];   // 1875 f32
  __shared__ _Float16 Wh[OUT_ * 32];     // 96 x (K padded 12->32)
  __shared__ _Float16 Yh[32 * 32];       // K(32) x N(32), pads zero
  __shared__ float    xs[C_ * V_];

  const int n      = blockIdx.x / (T_ / TBLK);
  const int tchunk = blockIdx.x % (T_ / TBLK);
  const int tid  = threadIdx.x;
  const int wave = tid >> 5, lane = tid & 31;
  const int hi = lane >> 4, lm = lane & 15;

  const float* Asrc = Asum + n * (3 * V_ * V_);
  for (int i = tid; i < 3 * V_ * V_; i += 256) As[i] = Asrc[i];
  for (int i = tid; i < OUT_ * 32; i += 256) {
    int o = i >> 5, k = i & 31;
    Wh[i] = (k < 12) ? (_Float16)W[o * 12 + k] : (_Float16)0.f;
  }
  __syncthreads();

  const float invs = rsqrtf(1.0f + 1e-5f);   // BN with running stats (0,1)

  for (int tt = 0; tt < TBLK; ++tt) {
    const int t = tchunk * TBLK + tt;
    for (int i = tid; i < C_ * V_; i += 256)
      xs[i] = x[((n * C_ + (i / V_)) * T_ + t) * V_ + (i % V_)];
    // prefetch next t-slice of x (global_prefetch_b8)
    if (tt + 1 < TBLK && tid < C_ * V_)
      __builtin_prefetch(&x[((n * C_ + (tid / V_)) * T_ + (t + 1)) * V_ + (tid % V_)], 0, 3);
    __syncthreads();

    // Y[ch=s*3+c, v] = sum_u Asum[s*25+v, u] * x[c, u] ; s=0 block == 300*I
    for (int i = tid; i < 32 * 32; i += 256) {
      int k = i >> 5, v = i & 31;
      float val = 0.f;
      if (k < 12 && v < V_) {
        int s = k / 3, c = k % 3;
        if (s == 0) {
          val = (float)T_ * xs[c * V_ + v];
        } else {
          const float* Ar = As + (s - 1) * V_ * V_ + v * V_;
          const float* xr = xs + c * V_;
          float acc = 0.f;
          #pragma unroll
          for (int u = 0; u < V_; ++u) acc += Ar[u] * xr[u];
          val = acc;
        }
      }
      Yh[i] = (_Float16)val;
    }
    __syncthreads();

    // out[96 x 25] = W(96x32) @ Y(32x32), 6 M-tiles x 2 N-tiles over 8 waves
    for (int job = wave; job < 12; job += 8) {
      const int mrow = (job >> 1) * 16;
      const int ncol = (job & 1) * 16;
      v16h a, b;
      // A fragment: 16-bit A 16x32 layout (VGPR0-3 K=0..7|8..15, VGPR4-7 K=16..23|24..31)
      const int arow = (mrow + lm) * 32;
      #pragma unroll
      for (int g = 0; g < 8; ++g)
        #pragma unroll
        for (int p = 0; p < 2; ++p) {
          int k = (g < 4) ? (2 * g + p + 8 * hi) : (16 + 2 * (g - 4) + p + 8 * hi);
          a[2 * g + p] = Wh[arow + k];
        }
      // B fragment: same K order as A (lanes 0-15 K=0..7,16..23; lanes 16-31 K=8..15,24..31)
      const int bn = ncol + lm;
      #pragma unroll
      for (int g = 0; g < 8; ++g)
        #pragma unroll
        for (int p = 0; p < 2; ++p) {
          int k = (g < 4) ? (2 * g + p + 8 * hi) : (16 + 2 * (g - 4) + p + 8 * hi);
          b[2 * g + p] = Yh[k * 32 + bn];
        }
      v8f acc = {};
      acc = __builtin_amdgcn_wmma_f32_16x16x32_f16(false, a, false, b,
                                                   (short)0, acc, false, false);
      const int v = ncol + lm;
      if (v < V_) {
        #pragma unroll
        for (int r = 0; r < 8; ++r) {
          int o = mrow + 8 * hi + r;
          float val = acc[r] + bias[o];
          val = val * (gamma[o] * invs) + beta[o];
          out[((n * OUT_ + o) * T_ + t) * V_ + v] = fmaxf(val, 0.f);
        }
      }
    }
    __syncthreads();
  }
}

extern "C" void kernel_launch(void* const* d_in, const int* in_sizes, int n_in,
                              void* d_out, int out_size, void* d_ws, size_t ws_size,
                              hipStream_t stream) {
  const float* x  = (const float*)d_in[0];   // (32,3,300,25)
  const float* xx = (const float*)d_in[1];   // (32,3,300,25,1)
  const float* W  = (const float*)d_in[2];   // (96,12)
  const float* b  = (const float*)d_in[3];   // (96,)
  const float* g  = (const float*)d_in[4];   // (96,)
  const float* be = (const float*)d_in[5];   // (96,)
  float* out  = (float*)d_out;               // (32,96,300,25)
  float* Asum = (float*)d_ws;                // 32*3*25*25 floats = 240 KB

  const int nz = N_ * 3 * V_ * V_;
  k_zero_ws<<<(nz + 255) / 256, 256, 0, stream>>>(Asum, nz);
  k_build_Asum<<<N_ * (T_ / TCH), 128, 0, stream>>>(xx, Asum);
  k_fused_out<<<N_ * (T_ / TBLK), 256, 0, stream>>>(x, W, b, g, be, Asum, out);
}
